// ProbeDecoder_16320875725328
// MI455X (gfx1250) — compile-verified, matrix operand order checked by reading, not run on hardware
//
#include <hip/hip_runtime.h>
#include <hip/hip_bf16.h>
#include <stdint.h>

// ---------------------------------------------------------------------------
// Problem constants (from reference)
// ---------------------------------------------------------------------------
#define N_SRC_    8192
#define N_Q_      16384
#define KNN_      8
#define N_EDGE_   (N_Q_*KNN_)      // 131072
#define NODE_DIM_ 256
#define EDGE_DIM_ 128
#define HIDDEN_   256
#define OUT_DIM_  4

// Stage bf16 X tiles via gfx1250 async global->LDS DMA (accepted by assembler).
#define ASYNC_LDS 1

typedef __bf16 bf16_t;
typedef __attribute__((ext_vector_type(16))) __bf16 v16bf;
typedef __attribute__((ext_vector_type(8)))  float  v8f;

__device__ __forceinline__ float gelu_exact(float x) {
    return 0.5f * x * (1.0f + erff(x * 0.7071067811865475f));
}

__device__ __forceinline__ void wait_asynccnt0() {
#if defined(__has_builtin) && __has_builtin(__builtin_amdgcn_s_wait_asynccnt)
    __builtin_amdgcn_s_wait_asynccnt(0);
#else
    asm volatile("s_wait_asynccnt 0x0" ::: "memory");
#endif
}

// ---------------------------------------------------------------------------
// kNN: brute-force top-8 over 8192 sources; positions staged in 96KB LDS
// ---------------------------------------------------------------------------
__global__ __launch_bounds__(256) void knn_kernel(
    const float* __restrict__ pos, const float* __restrict__ qpos,
    int* __restrict__ nn_idx, float* __restrict__ nn_dist)
{
    extern __shared__ float spos[];                       // N_SRC_*3 floats
    for (int i = threadIdx.x; i < N_SRC_*3; i += 256) spos[i] = pos[i];
    __syncthreads();

    const int q = blockIdx.x * 256 + threadIdx.x;
    const float qx = qpos[q*3+0], qy = qpos[q*3+1], qz = qpos[q*3+2];

    float bd[KNN_]; int bi[KNN_];
#pragma unroll
    for (int j = 0; j < KNN_; ++j) { bd[j] = 3.0e38f; bi[j] = 0; }

    for (int s = 0; s < N_SRC_; ++s) {
        float dx = qx - spos[s*3+0];
        float dy = qy - spos[s*3+1];
        float dz = qz - spos[s*3+2];
        float d2 = dx*dx + dy*dy + dz*dz;
        if (d2 < bd[KNN_-1]) {
            float vd = d2; int vi = s;
#pragma unroll
            for (int j = 0; j < KNN_; ++j) {
                if (vd < bd[j]) {
                    float td = bd[j]; int ti = bi[j];
                    bd[j] = vd; bi[j] = vi; vd = td; vi = ti;
                }
            }
        }
    }
#pragma unroll
    for (int j = 0; j < KNN_; ++j) {
        nn_idx [q*KNN_+j] = bi[j];
        nn_dist[q*KNN_+j] = sqrtf(fmaxf(bd[j], 0.0f));
    }
}

// ---------------------------------------------------------------------------
// probe_nodes[q] = mean of 8 gathered source node features
// ---------------------------------------------------------------------------
__global__ __launch_bounds__(256) void probe_init_kernel(
    const float* __restrict__ nodes, const int* __restrict__ nnIdx,
    float* __restrict__ pn)
{
    size_t t = (size_t)blockIdx.x * 256 + threadIdx.x;    // N_Q_*NODE_DIM_
    int q = (int)(t >> 8);
    int c = (int)(t & 255);
    float s = 0.f;
#pragma unroll
    for (int j = 0; j < KNN_; ++j)
        s += nodes[(size_t)nnIdx[q*KNN_+j]*NODE_DIM_ + c];
    pn[t] = s * 0.12499999984375f;                        // 1/(8 + 1e-8)
}

// ---------------------------------------------------------------------------
// colsum of edge-encoder W0 (all 8 RBF inputs equal the distance)
// ---------------------------------------------------------------------------
__global__ void colsum8_kernel(const float* __restrict__ W0, float* __restrict__ s)
{
    int n = threadIdx.x;                                  // 0..255
    float a = 0.f;
#pragma unroll
    for (int j = 0; j < KNN_; ++j) a += W0[j*HIDDEN_ + n];
    s[n] = a;
}

// h[e,n] = gelu(dist[e] * colsum(W0)[n] + b0[n])  -> bf16 hidden
__global__ __launch_bounds__(256) void rbf_hidden_kernel(
    const float* __restrict__ dist, const float* __restrict__ s,
    const float* __restrict__ b, bf16_t* __restrict__ h)
{
    size_t t = (size_t)blockIdx.x * 256 + threadIdx.x;    // N_EDGE_*HIDDEN_
    int e = (int)(t >> 8);
    int n = (int)(t & 255);
    float x = dist[e]*s[n] + b[n];
    h[t] = (bf16_t)gelu_exact(x);
}

// ---------------------------------------------------------------------------
// Repack fp32 W[K,N] -> bf16 in B-fragment order for v_wmma_f32_16x16x32_bf16
// dword index = ((kt*NT + nt)*32 + lane)*8 + r ; value packs {K=kb, K=kb+1}
// ---------------------------------------------------------------------------
__global__ __launch_bounds__(256) void repack_kernel(
    const float* __restrict__ W, uint32_t* __restrict__ Wp,
    int K, int N, int NT, int total)
{
    int t = blockIdx.x*256 + threadIdx.x;
    if (t >= total) return;
    int r     = t & 7;
    int lane  = (t >> 3) & 31;
    int tilei = t >> 8;
    int nt    = tilei % NT;
    int kt    = tilei / NT;
    int n     = nt*16 + (lane & 15);
    int k     = kt*32 + ((r < 4) ? 0 : 16) + (((lane >> 4) & 1) << 3) + ((r & 3) << 1);
    float lo = 0.f, hi = 0.f;
    if (n < N) { lo = W[(size_t)k*N + n]; hi = W[(size_t)(k+1)*N + n]; }
    union { bf16_t b[2]; uint32_t u; } cv;
    cv.b[0] = (bf16_t)lo; cv.b[1] = (bf16_t)hi;
    Wp[t] = cv.u;
}

// ---------------------------------------------------------------------------
// segment_sum over receivers: agg[q] = sum_j new_edges[q*8+j]
// ---------------------------------------------------------------------------
__global__ __launch_bounds__(256) void agg_kernel(
    const float* __restrict__ ne, float* __restrict__ agg)
{
    size_t t = (size_t)blockIdx.x*256 + threadIdx.x;      // N_Q_*EDGE_DIM_
    int q = (int)(t >> 7);
    int c = (int)(t & 127);
    float s = 0.f;
#pragma unroll
    for (int j = 0; j < KNN_; ++j)
        s += ne[(size_t)(q*KNN_+j)*EDGE_DIM_ + c];
    agg[t] = s;
}

// ---------------------------------------------------------------------------
// Fused WMMA GEMM, fully static tiling (no guards in the k-loop):
//   RW row-waves x (8/RW) col-waves, each wave owns exactly CT 16x16 tiles,
//   with (8/RW)*CT*16 == NDIM (padded to 16 for NDIM=4).
//   - up to 3 K-segments; per-segment row map: 0=identity 1=senders[row] 2=row>>3
//   - X staged to LDS bf16 [ROWS x 32]; bf16 inputs use async global->LDS DMA
//   - W pre-swizzled to exact B-fragment order (contiguous 32B load per lane)
// ---------------------------------------------------------------------------
template<int RW, int NDIM, bool XBF16, bool OUTBF16, bool ACT>
__global__ __launch_bounds__(256) void gemm_wmma(
    const void* __restrict__ X0, const void* __restrict__ X1, const void* __restrict__ X2,
    int ld0, int ld1, int ld2,
    int kend0, int kend1, int kend2,
    int mode0, int mode1, int mode2,
    const int* __restrict__ senders,
    const uint32_t* __restrict__ Wp, const float* __restrict__ bias,
    void* __restrict__ out, float* __restrict__ extra, int outmode,
    int Ktot)
{
    constexpr int ROWS = RW * 16;               // rows per block
    constexpr int COLW = 8 / RW;                // col-waves
    constexpr int NT   = (NDIM + 15) / 16;      // n-tiles total
    constexpr int CT   = NT / COLW;             // tiles per wave
    static_assert(COLW * CT == NT, "tile split must be exact");

    __shared__ bf16_t tile[ROWS * 40];          // ROWS x 32 k, stride 40
    __shared__ int    rowbase[3][ROWS];

    const int tid = threadIdx.x;
    const int blockRow = blockIdx.x * ROWS;

    for (int i = tid; i < 3 * ROWS; i += 256) {
        int s = i / ROWS, row = i % ROWS;
        int gr = blockRow + row;
        int mode = (s == 0) ? mode0 : (s == 1) ? mode1 : mode2;
        int ld   = (s == 0) ? ld0   : (s == 1) ? ld1   : ld2;
        int m = (mode == 1) ? senders[gr] : (mode == 2) ? (gr >> 3) : gr;
        rowbase[s][row] = m * ld;
    }

    const int lane = tid & 31;
    const int wid     = __builtin_amdgcn_readfirstlane(tid >> 5);   // scalar
    const int colwave = wid % COLW;
    const int rowwave = wid / COLW;

    v8f acc[CT];
#pragma unroll
    for (int c = 0; c < CT; ++c)
#pragma unroll
        for (int r = 0; r < 8; ++r) acc[c][r] = 0.0f;

    const int nKT = Ktot >> 5;
    const v16bf* wpv = (const v16bf*)Wp;

    for (int kt = 0; kt < nKT; ++kt) {
        const int k0 = kt << 5;
        int seg, ksub;
        const void* Xs;
        if (k0 < kend0)      { seg = 0; Xs = X0; ksub = k0; }
        else if (k0 < kend1) { seg = 1; Xs = X1; ksub = k0 - kend0; }
        else                 { seg = 2; Xs = X2; ksub = k0 - kend1; }

        __syncthreads();                  // protect LDS WAR from prev iter
        // ---- stage ROWS x 32 tile: 4-elem chunks, ROWS/32 chunks per thread
#pragma unroll
        for (int it = 0; it < ROWS / 32; ++it) {
            const int idx = tid + it * 256;           // chunk id
            const int row = idx >> 3;
            const int kk  = (idx & 7) << 2;
            const int base = rowbase[seg][row] + ksub + kk;
            bf16_t* dst = &tile[row*40 + kk];
            if (XBF16) {
                const bf16_t* xp = (const bf16_t*)Xs + base;
#if ASYNC_LDS
                const uint32_t loff  = (uint32_t)(uintptr_t)dst;      // LDS offset
                const uint64_t gaddr = (uint64_t)(uintptr_t)xp;
                asm volatile("global_load_async_to_lds_b64 %0, %1, off"
                             :: "v"(loff), "v"(gaddr) : "memory");
#else
#pragma unroll
                for (int i = 0; i < 4; ++i) dst[i] = xp[i];
#endif
            } else {
                const float* xp = (const float*)Xs + base;
#pragma unroll
                for (int i = 0; i < 4; ++i) dst[i] = (bf16_t)xp[i];
                __builtin_prefetch(xp + 32, 0, 1);
            }
        }
        if (XBF16) {
#if ASYNC_LDS
            wait_asynccnt0();
#endif
        }
        __syncthreads();

        // ---- A fragment from LDS (16-bit A 16x32 layout, wave32)
        union { v16bf v; uint32_t u[8]; } a;
        const int m = (lane & 15) + (rowwave << 4);
#pragma unroll
        for (int r = 0; r < 8; ++r) {
            const int kk = ((r < 4) ? 0 : 16) + (((lane >> 4) & 1) << 3) + ((r & 3) << 1);
            a.u[r] = *(const uint32_t*)&tile[m*40 + kk];
        }

        // ---- B fragments (contiguous 32B per lane), then WMMAs back-to-back
        v16bf bf[CT];
#pragma unroll
        for (int c = 0; c < CT; ++c)
            bf[c] = wpv[(size_t)(kt*NT + colwave*CT + c)*32 + lane];
#pragma unroll
        for (int c = 0; c < CT; ++c)
            acc[c] = __builtin_amdgcn_wmma_f32_16x16x32_bf16(
                false, a.v, false, bf[c], (short)0, acc[c], false, false);
    }

    // ---- Epilogue: lane holds col N=lane&15, rows v + 8*(lane>=16)
    const int col0  = lane & 15;
    const int rbase = blockRow + (rowwave << 4) + (((lane >> 4) & 1) << 3);
#pragma unroll
    for (int c = 0; c < CT; ++c) {
        const int n = (colwave*CT + c) * 16 + col0;
        if ((NDIM & 15) != 0 && n >= NDIM) continue;      // folds away if NDIM%16==0
        const float bia = bias ? bias[n] : 0.0f;
#pragma unroll
        for (int r = 0; r < 8; ++r) {
            const int gr = rbase + r;
            float v = acc[c][r] + bia;
            if (ACT) v = gelu_exact(v);
            const size_t o = (size_t)gr * NDIM + n;
            if (extra) extra[o] += v;
            if (OUTBF16) {
                ((bf16_t*)out)[o] = (bf16_t)v;
            } else {
                float* op = (float*)out;
                op[o] = outmode ? (op[o] + v) : v;
            }
        }
    }
}

// ---------------------------------------------------------------------------
static inline size_t repack_dwords(int K, int N) {
    int NT = (N + 15) / 16;
    return (size_t)(K/32) * NT * 256;
}

extern "C" void kernel_launch(void* const* d_in, const int* in_sizes, int n_in,
                              void* d_out, int out_size, void* d_ws, size_t ws_size,
                              hipStream_t stream)
{
    (void)in_sizes; (void)n_in; (void)out_size; (void)ws_size;

    // setup_inputs() insertion order; params pytree flattened in declaration order
    const float* nodes = (const float*)d_in[0];
    const float* pos   = (const float*)d_in[1];
    const float* qpos  = (const float*)d_in[2];
    const float* eeW0  = (const float*)d_in[3];
    const float* eeB0  = (const float*)d_in[4];
    const float* eeW1  = (const float*)d_in[5];
    const float* eeB1  = (const float*)d_in[6];
    const float* eW0[2] = {(const float*)d_in[7],  (const float*)d_in[15]};
    const float* eB0[2] = {(const float*)d_in[8],  (const float*)d_in[16]};
    const float* eW1[2] = {(const float*)d_in[9],  (const float*)d_in[17]};
    const float* eB1[2] = {(const float*)d_in[10], (const float*)d_in[18]};
    const float* nW0[2] = {(const float*)d_in[11], (const float*)d_in[19]};
    const float* nB0[2] = {(const float*)d_in[12], (const float*)d_in[20]};
    const float* nW1[2] = {(const float*)d_in[13], (const float*)d_in[21]};
    const float* nB1[2] = {(const float*)d_in[14], (const float*)d_in[22]};
    const float* oW0 = (const float*)d_in[23];
    const float* oB0 = (const float*)d_in[24];
    const float* oW1 = (const float*)d_in[25];
    const float* oB1 = (const float*)d_in[26];
    const float* oW2 = (const float*)d_in[27];
    const float* oB2 = (const float*)d_in[28];

    char* ws = (char*)d_ws;
    size_t off = 0;
    auto alloc = [&](size_t bytes) -> char* {
        char* p = ws + off;
        off = (off + bytes + 255) & ~(size_t)255;
        return p;
    };
    float*  colS   = (float*)  alloc((size_t)HIDDEN_*4);
    int*    nnIdx  = (int*)    alloc((size_t)N_EDGE_*4);
    float*  nnDist = (float*)  alloc((size_t)N_EDGE_*4);
    float*  edges  = (float*)  alloc((size_t)N_EDGE_*EDGE_DIM_*4);
    float*  nedges = (float*)  alloc((size_t)N_EDGE_*EDGE_DIM_*4);
    bf16_t* hidden = (bf16_t*) alloc((size_t)N_EDGE_*HIDDEN_*2);
    float*  pn     = (float*)  alloc((size_t)N_Q_*NODE_DIM_*4);
    float*  agg    = (float*)  alloc((size_t)N_Q_*EDGE_DIM_*4);

    uint32_t* WpEE    = (uint32_t*)alloc(repack_dwords(256,128)*4);
    uint32_t* WpE0[2] = {(uint32_t*)alloc(repack_dwords(640,256)*4),
                         (uint32_t*)alloc(repack_dwords(640,256)*4)};
    uint32_t* WpE1[2] = {(uint32_t*)alloc(repack_dwords(256,128)*4),
                         (uint32_t*)alloc(repack_dwords(256,128)*4)};
    uint32_t* WpN0[2] = {(uint32_t*)alloc(repack_dwords(384,256)*4),
                         (uint32_t*)alloc(repack_dwords(384,256)*4)};
    uint32_t* WpN1[2] = {(uint32_t*)alloc(repack_dwords(256,256)*4),
                         (uint32_t*)alloc(repack_dwords(256,256)*4)};
    uint32_t* WpO0 = (uint32_t*)alloc(repack_dwords(256,256)*4);
    uint32_t* WpO1 = (uint32_t*)alloc(repack_dwords(256,256)*4);
    uint32_t* WpO2 = (uint32_t*)alloc(repack_dwords(256,4)*4);

    auto repack = [&](const float* W, uint32_t* Wp, int K, int N) {
        int NT = (N + 15) / 16;
        int total = (int)repack_dwords(K, N);
        repack_kernel<<<(total + 255)/256, 256, 0, stream>>>(W, Wp, K, N, NT, total);
    };

    // ---- stage 1: kNN graph + probe init + degenerate RBF layer ----
    knn_kernel<<<N_Q_/256, 256, N_SRC_*3*sizeof(float), stream>>>(pos, qpos, nnIdx, nnDist);
    colsum8_kernel<<<1, HIDDEN_, 0, stream>>>(eeW0, colS);
    probe_init_kernel<<<(N_Q_*NODE_DIM_)/256, 256, 0, stream>>>(nodes, nnIdx, pn);
    rbf_hidden_kernel<<<(int)(((size_t)N_EDGE_*HIDDEN_)/256), 256, 0, stream>>>(nnDist, colS, eeB0, hidden);

    // ---- weight repack (per-launch, deterministic) ----
    repack(eeW1, WpEE, 256, 128);
    for (int l = 0; l < 2; ++l) {
        repack(eW0[l], WpE0[l], 640, 256);
        repack(eW1[l], WpE1[l], 256, 128);
        repack(nW0[l], WpN0[l], 384, 256);
        repack(nW1[l], WpN1[l], 256, 256);
    }
    repack(oW0, WpO0, 256, 256);
    repack(oW1, WpO1, 256, 256);
    repack(oW2, WpO2, 256, 4);

    // ---- edge encoder layer 2: hidden(bf16) @ W1 + b1 -> edges (f32) ----
    gemm_wmma<4,EDGE_DIM_,true,false,false><<<N_EDGE_/64, 256, 0, stream>>>(
        hidden, hidden, hidden, HIDDEN_, 0, 0, 256, 256, 256, 0, 0, 0,
        nnIdx, WpEE, eeB1, edges, nullptr, 0, 256);

    // ---- 2 message-passing layers ----
    for (int l = 0; l < 2; ++l) {
        // edge_mlp L1: [pn[senders] | pn[recv] | edges] (K=640) -> hidden, GELU
        gemm_wmma<2,HIDDEN_,false,true,true><<<N_EDGE_/32, 256, 0, stream>>>(
            pn, pn, edges, NODE_DIM_, NODE_DIM_, EDGE_DIM_,
            256, 512, 640, 1, 2, 0,
            nnIdx, WpE0[l], eB0[l], hidden, nullptr, 0, 640);
        // edge_mlp L2: hidden -> new_edges ; fused edges += new_edges
        gemm_wmma<4,EDGE_DIM_,true,false,false><<<N_EDGE_/64, 256, 0, stream>>>(
            hidden, hidden, hidden, HIDDEN_, 0, 0, 256, 256, 256, 0, 0, 0,
            nnIdx, WpE1[l], eB1[l], nedges, edges, 0, 256);
        // aggregated[q] = sum_j new_edges[q*8+j]
        agg_kernel<<<(int)(((size_t)N_Q_*EDGE_DIM_)/256), 256, 0, stream>>>(nedges, agg);
        // node_mlp L1: [pn | agg] (K=384) -> hidden, GELU
        gemm_wmma<2,HIDDEN_,false,true,true><<<N_Q_/32, 256, 0, stream>>>(
            pn, agg, agg, NODE_DIM_, EDGE_DIM_, 0,
            256, 384, 384, 0, 0, 0,
            nnIdx, WpN0[l], nB0[l], hidden, nullptr, 0, 384);
        // node_mlp L2: hidden -> pn (residual add, outmode=1)
        gemm_wmma<2,NODE_DIM_,true,false,false><<<N_Q_/32, 256, 0, stream>>>(
            hidden, hidden, hidden, HIDDEN_, 0, 0, 256, 256, 256, 0, 0, 0,
            nnIdx, WpN1[l], nB1[l], pn, nullptr, 1, 256);
    }

    // ---- output MLP: 256 -> 256 -> 256 -> 4 ----
    gemm_wmma<2,HIDDEN_,false,true,true><<<N_Q_/32, 256, 0, stream>>>(
        pn, pn, pn, NODE_DIM_, 0, 0, 256, 256, 256, 0, 0, 0,
        nnIdx, WpO0, oB0, hidden, nullptr, 0, 256);
    gemm_wmma<2,HIDDEN_,true,true,true><<<N_Q_/32, 256, 0, stream>>>(  // in-place, per-row safe
        hidden, hidden, hidden, HIDDEN_, 0, 0, 256, 256, 256, 0, 0, 0,
        nnIdx, WpO1, oB1, hidden, nullptr, 0, 256);
    gemm_wmma<8,OUT_DIM_,true,false,false><<<N_Q_/128, 256, 0, stream>>>(
        hidden, hidden, hidden, HIDDEN_, 0, 0, 256, 256, 256, 0, 0, 0,
        nnIdx, WpO2, oB2, d_out, nullptr, 0, 256);
}
